// HuGNNBackbone_5755256176701
// MI455X (gfx1250) — matrix operand order repaired and checked by reading, not compiled, not run on hardware
//
#include <hip/hip_runtime.h>

typedef __bf16 bf16;
typedef __attribute__((ext_vector_type(16))) __bf16 v16bf;
typedef __attribute__((ext_vector_type(8)))  __bf16 bf16x8;
typedef __attribute__((ext_vector_type(4)))  __bf16 bf16x4;
typedef __attribute__((ext_vector_type(8)))  float  v8f;
typedef __attribute__((ext_vector_type(4)))  float  f4;

#define DV   300   // real emb dim
#define DP   320   // padded emb dim (mult of 32)
#define HV   600   // real hidden
#define HP   640   // padded hidden (mult of 32)
#define LNUM 5
#define BN_EPS 1e-5f
#define C4   (DV / 4)   // 75 float4 chunks per row

__device__ __forceinline__ void atomic_add_f32(float* p, float v) {
  __hip_atomic_fetch_add(p, v, __ATOMIC_RELAXED, __HIP_MEMORY_SCOPE_AGENT);
}

// ---------------- weight conversion: f32 [K][N] -> padded TRANSPOSED bf16 [N][K] ----------------
__global__ void k_w1_to_bf16t(const float* __restrict__ W1, bf16* __restrict__ W1t) {
  int idx = blockIdx.x * blockDim.x + threadIdx.x;
  const int total = LNUM * HP * DP;
  if (idx >= total) return;
  int l = idx / (HP * DP);
  int r = idx % (HP * DP);
  int n = r / DP, k = r % DP;
  float v = (k < DV && n < HV) ? W1[((size_t)l * DV + k) * HV + n] : 0.f;
  W1t[idx] = (bf16)v;
}
__global__ void k_w2_to_bf16t(const float* __restrict__ W2, bf16* __restrict__ W2t) {
  int idx = blockIdx.x * blockDim.x + threadIdx.x;
  const int total = LNUM * DP * HP;
  if (idx >= total) return;
  int l = idx / (DP * HP);
  int r = idx % (DP * HP);
  int n = r / HP, k = r % HP;
  float v = (n < DV && k < HV) ? W2[((size_t)l * HV + k) * DV + n] : 0.f;
  W2t[idx] = (bf16)v;
}

// ---------------- input atom embedding (float4) ----------------
__global__ void k_embed(const int* __restrict__ x, const float* __restrict__ emb1,
                        const float* __restrict__ emb2, float* __restrict__ h, int n) {
  int idx = blockIdx.x * blockDim.x + threadIdx.x;
  if (idx >= n * C4) return;
  int i = idx / C4, q = idx % C4;
  f4 a = *(const f4*)(emb1 + (size_t)x[2 * i] * DV + q * 4);
  f4 b = *(const f4*)(emb2 + (size_t)x[2 * i + 1] * DV + q * 4);
  *(f4*)(h + (size_t)i * DV + q * 4) = a + b;
}

// ---------------- aggr = h + self-loop edge emb (float4) ----------------
__global__ void k_aggr_init(const float* __restrict__ h, const float* __restrict__ sl1,
                            const float* __restrict__ sl2, float* __restrict__ aggr, int n) {
  int idx = blockIdx.x * blockDim.x + threadIdx.x;
  if (idx >= n * C4) return;
  int q = idx % C4;
  f4 hv = *(const f4*)(h + (size_t)idx * 4);
  f4 s1 = *(const f4*)(sl1 + q * 4);
  f4 s2 = *(const f4*)(sl2 + q * 4);
  *(f4*)(aggr + (size_t)idx * 4) = hv + s1 + s2;
}

// ---------------- edge gather + scatter-add (float4 gather, f32 HW atomics) ----------------
__global__ void k_scatter(const float* __restrict__ h, const int* __restrict__ ei,
                          const int* __restrict__ ea, const float* __restrict__ ee1l,
                          const float* __restrict__ ee2l, float* __restrict__ aggr, int E) {
  int idx = blockIdx.x * blockDim.x + threadIdx.x;
  if (idx >= E * C4) return;
  int e = idx / C4, q = idx % C4;
  int s  = ei[e], d = ei[E + e];
  int bt = ea[2 * e], bd = ea[2 * e + 1];
  f4 hv = *(const f4*)(h + (size_t)s * DV + q * 4);
  f4 e1 = *(const f4*)(ee1l + bt * DV + q * 4);
  f4 e2 = *(const f4*)(ee2l + bd * DV + q * 4);
  f4 val = hv + e1 + e2;
  float* dst = aggr + (size_t)d * DV + q * 4;
#pragma unroll
  for (int t = 0; t < 4; ++t) atomic_add_f32(dst + t, val[t]);
}

// ---------------- (1+eps)*aggr -> padded bf16 (4-wide) ----------------
__global__ void k_scale_pad(const float* __restrict__ aggr, const float* __restrict__ eps,
                            int l, bf16* __restrict__ z0, int n) {
  int idx = blockIdx.x * blockDim.x + threadIdx.x;
  if (idx >= n * (DP / 4)) return;
  int i = idx / (DP / 4), q = idx % (DP / 4);
  int c0 = q * 4;                    // 300 % 4 == 0: a chunk is all-real or all-pad
  bf16x4 o;
  if (c0 < DV) {
    float sc = 1.f + eps[l];
    f4 v = *(const f4*)(aggr + (size_t)i * DV + c0);
#pragma unroll
    for (int t = 0; t < 4; ++t) o[t] = (bf16)(sc * v[t]);
  } else {
#pragma unroll
    for (int t = 0; t < 4; ++t) o[t] = (bf16)0.f;
  }
  *(bf16x4*)(z0 + (size_t)i * DP + c0) = o;
}

// ---------------- bf16 WMMA GEMM: 2 M-tiles x 4 N-tiles per wave ----------------
// A:  [32*MPairs, lda] bf16 row-major.
// Bt: [Ncols, ldk] bf16 row-major = TRANSPOSED weights (col n's K-vector contiguous).
// All 12 fragment loads of a k-step are issued before the 8 WMMAs so the
// scheduler can stagger s_wait_loadcnt and overlap VMEM with matrix math.
union frag16 { v16bf v; bf16x8 h[2]; };

__global__ __launch_bounds__(256)
void k_gemm_bf16(const bf16* __restrict__ A, int lda,
                 const bf16* __restrict__ Bt, int ldk,
                 const float* __restrict__ bias, int nbias,
                 bf16* __restrict__ Cbf, float* __restrict__ Cf, int ldc,
                 int MPairs, int NtGroups, int Ksteps, int relu_bf16) {
  const int wave = threadIdx.x >> 5;
  const int lane = threadIdx.x & 31;
  const int wg   = blockIdx.x * (blockDim.x >> 5) + wave;
  if (wg >= MPairs * NtGroups) return;   // uniform per wave: EXEC stays all-ones
  const int tp   = wg / NtGroups;        // M-tile pair
  const int tg   = wg % NtGroups;        // group of 4 N-tiles
  const int m    = lane & 15;
  const int half = lane >> 4;

  v8f acc[2][4];
#pragma unroll
  for (int mi = 0; mi < 2; ++mi)
#pragma unroll
    for (int j = 0; j < 4; ++j)
#pragma unroll
      for (int i = 0; i < 8; ++i) acc[mi][j][i] = 0.f;

  // A 16x32 bf16 fragment: half 0 -> K {k0..k0+7, k0+16..k0+23}; half 1 -> +8
  const bf16* arow0 = A + (size_t)(tp * 32 + m) * lda + half * 8;
  const bf16* arow1 = arow0 + (size_t)16 * lda;
  // B 32x16 fragment from transposed weights: half 0 -> K {k0..k0+7, k0+16..k0+23}; half 1 -> +8
  int cn[4];
  const bf16* brow[4];
#pragma unroll
  for (int j = 0; j < 4; ++j) {
    cn[j]   = (tg * 4 + j) * 16 + m;
    brow[j] = Bt + (size_t)cn[j] * ldk + half * 8;
  }

  for (int ks = 0; ks < Ksteps; ++ks) {
    const int k0 = ks * 32;
    frag16 a0, a1, b[4];
    a0.h[0] = *reinterpret_cast<const bf16x8*>(arow0 + k0);
    a0.h[1] = *reinterpret_cast<const bf16x8*>(arow0 + k0 + 16);
    a1.h[0] = *reinterpret_cast<const bf16x8*>(arow1 + k0);
    a1.h[1] = *reinterpret_cast<const bf16x8*>(arow1 + k0 + 16);
#pragma unroll
    for (int j = 0; j < 4; ++j) {
      b[j].h[0] = *reinterpret_cast<const bf16x8*>(brow[j] + k0);
      b[j].h[1] = *reinterpret_cast<const bf16x8*>(brow[j] + k0 + 16);
    }
#pragma unroll
    for (int j = 0; j < 4; ++j)
      acc[0][j] = __builtin_amdgcn_wmma_f32_16x16x32_bf16(false, a0.v, false, b[j].v,
                                                          (short)0, acc[0][j], false, false);
#pragma unroll
    for (int j = 0; j < 4; ++j)
      acc[1][j] = __builtin_amdgcn_wmma_f32_16x16x32_bf16(false, a1.v, false, b[j].v,
                                                          (short)0, acc[1][j], false, false);
  }

#pragma unroll
  for (int mi = 0; mi < 2; ++mi) {
    const int rowbase = tp * 32 + mi * 16 + half * 8;
#pragma unroll
    for (int j = 0; j < 4; ++j) {
      float bv = 0.f;
      if (bias != nullptr && cn[j] < nbias) bv = bias[cn[j]];
      if (relu_bf16) {
#pragma unroll
        for (int v = 0; v < 8; ++v) {      // C layout: VGPR v -> M = v + 8*half
          float val = fmaxf(acc[mi][j][v] + bv, 0.f);
          Cbf[(size_t)(rowbase + v) * ldc + cn[j]] = (bf16)val;
        }
      } else {
#pragma unroll
        for (int v = 0; v < 8; ++v) {
          Cf[(size_t)(rowbase + v) * ldc + cn[j]] = acc[mi][j][v] + bv;
        }
      }
    }
  }
}

// ---------------- batch-norm ----------------
__global__ void k_zero_stats(float* __restrict__ a, float* __restrict__ b) {
  int t = threadIdx.x;
  if (t < DV) { a[t] = 0.f; b[t] = 0.f; }
}
__global__ __launch_bounds__(320)
void k_bn_stats(const float* __restrict__ zf, float* __restrict__ sum,
                float* __restrict__ sumsq, int n) {
  int c = threadIdx.x;
  if (c >= DV) return;
  int row0 = blockIdx.x * 128;
  int rows = (n - row0 < 128) ? (n - row0) : 128;
  float s = 0.f, ss = 0.f;
  for (int r = 0; r < rows; ++r) {
    float v = zf[(size_t)(row0 + r) * DP + c];
    s += v; ss += v * v;
  }
  atomic_add_f32(&sum[c], s);
  atomic_add_f32(&sumsq[c], ss);
}
__global__ void k_bn_finalize(const float* __restrict__ sum, const float* __restrict__ sumsq,
                              const float* __restrict__ gamma_l, float* __restrict__ mean,
                              float* __restrict__ isg, int n) {
  int c = threadIdx.x;
  if (c >= DV) return;
  float inv_n = 1.f / (float)n;
  float mu = sum[c] * inv_n;
  float var = sumsq[c] * inv_n - mu * mu;
  mean[c] = mu;
  isg[c]  = rsqrtf(var + BN_EPS) * gamma_l[c];
}
__global__ void k_bn_apply(const float* __restrict__ zf, const float* __restrict__ mean,
                           const float* __restrict__ isg, const float* __restrict__ beta_l,
                           float* __restrict__ out, int n, int do_relu) {
  int idx = blockIdx.x * blockDim.x + threadIdx.x;
  if (idx >= n * C4) return;
  int i = idx / C4, q = idx % C4;
  f4 z  = *(const f4*)(zf + (size_t)i * DP + q * 4);
  f4 mu = *(const f4*)(mean + q * 4);
  f4 g  = *(const f4*)(isg + q * 4);
  f4 be = *(const f4*)(beta_l + q * 4);
  f4 v  = (z - mu) * g + be;
  if (do_relu) {
#pragma unroll
    for (int t = 0; t < 4; ++t) v[t] = fmaxf(v[t], 0.f);
  }
  *(f4*)(out + (size_t)i * DV + q * 4) = v;
}

static inline unsigned gdim(long n, int tb) { return (unsigned)((n + tb - 1) / tb); }

extern "C" void kernel_launch(void* const* d_in, const int* in_sizes, int n_in,
                              void* d_out, int out_size, void* d_ws, size_t ws_size,
                              hipStream_t stream) {
  const int*   x     = (const int*)d_in[0];
  const int*   ei    = (const int*)d_in[1];
  const int*   ea    = (const int*)d_in[2];
  const float* emb1  = (const float*)d_in[3];
  const float* emb2  = (const float*)d_in[4];
  const float* ee1   = (const float*)d_in[5];
  const float* ee2   = (const float*)d_in[6];
  const float* W1    = (const float*)d_in[7];
  const float* b1    = (const float*)d_in[8];
  const float* W2    = (const float*)d_in[9];
  const float* b2    = (const float*)d_in[10];
  const float* eps   = (const float*)d_in[11];
  const float* gamma = (const float*)d_in[12];
  const float* beta  = (const float*)d_in[13];
  (void)n_in; (void)out_size; (void)ws_size;

  const int N = in_sizes[0] / 2;   // nodes
  const int E = in_sizes[2] / 2;   // edges

  // --- carve workspace ---
  char* base = (char*)d_ws;
  size_t off = 0;
  auto carve = [&](size_t bytes) -> char* {
    char* p = base + off;
    off += (bytes + 255) & ~(size_t)255;
    return p;
  };
  float* h    = (float*)carve((size_t)N * DV * 4);
  float* aggr = (float*)carve((size_t)N * DV * 4);
  bf16*  z0   = (bf16*) carve((size_t)N * DP * 2);
  bf16*  h1   = (bf16*) carve((size_t)N * HP * 2);
  float* zf   = (float*)carve((size_t)N * DP * 4);
  bf16*  W1t  = (bf16*) carve((size_t)LNUM * HP * DP * 2);  // [l][n=640][k=320]
  bf16*  W2t  = (bf16*) carve((size_t)LNUM * DP * HP * 2);  // [l][n=320][k=640]
  float* bsum = (float*)carve(DV * 4);
  float* bsq  = (float*)carve(DV * 4);
  float* bmu  = (float*)carve(DV * 4);
  float* bisg = (float*)carve(DV * 4);

  const int TB = 256;

  k_w1_to_bf16t<<<gdim((long)LNUM * HP * DP, TB), TB, 0, stream>>>(W1, W1t);
  k_w2_to_bf16t<<<gdim((long)LNUM * DP * HP, TB), TB, 0, stream>>>(W2, W2t);
  k_embed<<<gdim((long)N * C4, TB), TB, 0, stream>>>(x, emb1, emb2, h, N);

  const int MPairs = N / 32;  // N = 100000: 32 | N  -> 3125 pairs of 16-row tiles

  for (int l = 0; l < LNUM; ++l) {
    const float* ee1l = ee1 + (size_t)l * 6 * DV;
    const float* ee2l = ee2 + (size_t)l * 3 * DV;

    k_aggr_init<<<gdim((long)N * C4, TB), TB, 0, stream>>>(h, ee1l + 4 * DV, ee2l, aggr, N);
    k_scatter<<<gdim((long)E * C4, TB), TB, 0, stream>>>(h, ei, ea, ee1l, ee2l, aggr, E);
    k_scale_pad<<<gdim((long)N * (DP / 4), TB), TB, 0, stream>>>(aggr, eps, l, z0, N);

    { // GEMM1: [N,320]bf16 x [320,640]bf16 -> relu -> bf16 h1   (Ntiles=40 -> 10 groups)
      int NtG = (HP / 16) / 4, Ks = DP / 32;
      long wgroups = (long)MPairs * NtG;
      k_gemm_bf16<<<gdim(wgroups, 8), 256, 0, stream>>>(
          z0, DP, W1t + (size_t)l * HP * DP, DP,
          b1 + (size_t)l * HV, HV, h1, nullptr, HP, MPairs, NtG, Ks, 1);
    }
    { // GEMM2: [N,640]bf16 x [640,320]bf16 -> f32 zf   (Ntiles=20 -> 5 groups)
      int NtG = (DP / 16) / 4, Ks = HP / 32;
      long wgroups = (long)MPairs * NtG;
      k_gemm_bf16<<<gdim(wgroups, 8), 256, 0, stream>>>(
          h1, HP, W2t + (size_t)l * DP * HP, HP,
          b2 + (size_t)l * DV, DV, nullptr, zf, DP, MPairs, NtG, Ks, 0);
    }

    k_zero_stats<<<1, 512, 0, stream>>>(bsum, bsq);
    k_bn_stats<<<gdim(N, 128), 320, 0, stream>>>(zf, bsum, bsq, N);
    k_bn_finalize<<<1, 320, 0, stream>>>(bsum, bsq, gamma + (size_t)l * DV, bmu, bisg, N);

    float* outp = (l == LNUM - 1) ? (float*)d_out : h;
    k_bn_apply<<<gdim((long)N * C4, TB), TB, 0, stream>>>(
        zf, bmu, bisg, beta + (size_t)l * DV, outp, N, (l < LNUM - 1) ? 1 : 0);
  }
}
